// Fusion_64802466562429
// MI455X (gfx1250) — matrix operand order repaired and checked by reading, not compiled, not run on hardware
//
#include <hip/hip_runtime.h>
#include <hip/hip_bf16.h>
#include <math.h>

typedef __attribute__((ext_vector_type(16))) _Float16 v16h;
typedef __attribute__((ext_vector_type(8)))  float    v8f;

#define LOW_H 128
#define LOW_W 240
#define LOW_HW (LOW_H * LOW_W)
#define HI_H 512
#define HI_W 960
#define HI_HW (HI_H * HI_W)
#define NB 2

enum { ACT_NONE = 0, ACT_RELU = 1, ACT_MISH = 2, ACT_SIGMOID = 3 };

__device__ __forceinline__ float act_apply(float v, int act) {
    if (act == ACT_RELU) return fmaxf(v, 0.f);
    if (act == ACT_MISH) {
        float sp = (v > 20.f) ? v : log1pf(__expf(v));
        return v * tanhf(sp);
    }
    if (act == ACT_SIGMOID) return 1.f / (1.f + __expf(-v));
    return v;
}

// ---------------------------------------------------------------------------
// f32 -> f16 cast (inputs that feed WMMA convs)
// ---------------------------------------------------------------------------
__global__ void cast_f32_f16(const float* __restrict__ in, _Float16* __restrict__ out,
                             int n)
{
    int i = blockIdx.x * blockDim.x + threadIdx.x;
    if (i < n) out[i] = (_Float16)in[i];
}

// ---------------------------------------------------------------------------
// Repack conv weights (Cout, Ktot) -> WMMA A-fragment order, f16, zero padded
// to CoutP=16*nmt (nmt even), KtotP=32*nkt.  Element (mt,kt,lane,h) holds
//   w[co*Ktot + k],  co = mt*16 + (lane&15),
//   k = kt*32 + (lane>=16 ? 8 : 0) + (h<8 ? h : h+8)
// stored flat at ((mt*nkt + kt)*32 + lane)*16 + h  (32B-aligned per lane).
// ---------------------------------------------------------------------------
__global__ void pack_weights(const float* __restrict__ w, _Float16* __restrict__ wp,
                             int Cout, int Ktot, int nmt, int nkt)
{
    int idx = blockIdx.x * blockDim.x + threadIdx.x;
    int total = nmt * nkt * 512;
    if (idx >= total) return;
    int h    = idx & 15;
    int lane = (idx >> 4) & 31;
    int t    = idx >> 9;             // mt*nkt + kt
    int kt   = t % nkt;
    int mt   = t / nkt;
    int co   = mt * 16 + (lane & 15);
    int k    = kt * 32 + ((lane >> 4) ? 8 : 0) + ((h < 8) ? h : h + 8);
    float v  = (co < Cout && k < Ktot) ? w[(long)co * Ktot + k] : 0.f;
    wp[idx]  = (_Float16)v;
}

// ---------------------------------------------------------------------------
// Implicit-GEMM conv via WMMA f16 (fp32 accumulate), f16 activations.
// 4 waves/block; each wave computes a 32(Cout) x 16(pixel) tile: the costly
// im2col B gather is built once per K-step and fed to TWO WMMAs (two packed
// A tiles), halving B traffic per output element.
// ---------------------------------------------------------------------------
template <int KS>
__global__ void __launch_bounds__(128)
conv2d_wmma(const _Float16* __restrict__ in, int inC, long inBS,
            const _Float16* __restrict__ wp, int nkt,
            const float* __restrict__ bias,
            _Float16* __restrict__ out, int Cout, long outBS, int coOff,
            const _Float16* __restrict__ res, long resBS,
            int H, int W, int pad, int act, int inRelu)
{
    constexpr int KK = KS * KS;
    const int lane = threadIdx.x & 31;
    const int wv   = threadIdx.x >> 5;
    const int HWl  = H * W;
    const int n0   = (blockIdx.x * 4 + wv) * 16;
    const int m0   = blockIdx.y * 32;
    const int b    = blockIdx.z;
    const int Ktot = inC * KK;

    const int half = lane >> 4;
    const int l15  = lane & 15;

    const int  pix   = n0 + l15;
    const bool pixOK = (pix < HWl);
    const int  pixc  = pixOK ? pix : (HWl - 1);
    const int  py    = pixc / W;
    const int  px    = pixc - py * W;

    const _Float16* inB   = in + (long)b * inBS;
    const _Float16* wrow0 = wp + ((long)(m0 >> 4) * nkt * 32 + lane) * 16;
    const _Float16* wrow1 = wrow0 + (long)nkt * 512;
    const int       kb0   = half ? 8 : 0;

    v8f acc0 = {0.f, 0.f, 0.f, 0.f, 0.f, 0.f, 0.f, 0.f};
    v8f acc1 = {0.f, 0.f, 0.f, 0.f, 0.f, 0.f, 0.f, 0.f};

    for (int kt = 0; kt < nkt; ++kt) {
        v16h a0 = *reinterpret_cast<const v16h*>(wrow0 + (long)kt * 512);
        v16h a1 = *reinterpret_cast<const v16h*>(wrow1 + (long)kt * 512);
        v16h bfrag;
        const int kb = kt * 32 + kb0;
#pragma unroll
        for (int h = 0; h < 16; ++h) {
            int  k   = kb + ((h < 8) ? h : h + 8);
            int  kc  = (k < Ktot) ? k : (Ktot - 1);
            int  ci  = kc / KK;
            int  rem = kc - ci * KK;
            int  ky  = rem / KS;
            int  kx  = rem - ky * KS;
            int  yy  = py + ky - pad;
            int  xx  = px + kx - pad;
            bool ok  = pixOK && (k < Ktot) &&
                       (yy >= 0) && (yy < H) && (xx >= 0) && (xx < W);
            int  yyc = yy < 0 ? 0 : (yy >= H ? H - 1 : yy);
            int  xxc = xx < 0 ? 0 : (xx >= W ? W - 1 : xx);
            _Float16 v = inB[(long)ci * HWl + yyc * W + xxc];  // always valid addr
            if (inRelu) v = (v > (_Float16)0.f) ? v : (_Float16)0.f;
            bfrag[h] = ok ? v : (_Float16)0.f;                  // cndmask, no branch
        }
        acc0 = __builtin_amdgcn_wmma_f32_16x16x32_f16(false, a0, false, bfrag,
                                                      (short)0, acc0, false, false);
        acc1 = __builtin_amdgcn_wmma_f32_16x16x32_f16(false, a1, false, bfrag,
                                                      (short)0, acc1, false, false);
    }

    if (!pixOK) return;
    _Float16*       outB = out + (long)b * outBS;
    const _Float16* resB = res ? (res + (long)b * resBS) : nullptr;
#pragma unroll
    for (int r = 0; r < 8; ++r) {
        int co0 = m0 + r + (half ? 8 : 0);
        int co1 = co0 + 16;
        if (co0 < Cout) {
            float v = acc0[r] + bias[co0];
            v = act_apply(v, act);
            if (resB) v += (float)resB[(long)co0 * HWl + pix];
            outB[(long)(co0 + coOff) * HWl + pix] = (_Float16)v;
        }
        if (co1 < Cout) {
            float v = acc1[r] + bias[co1];
            v = act_apply(v, act);
            if (resB) v += (float)resB[(long)co1 * HWl + pix];
            outB[(long)(co1 + coOff) * HWl + pix] = (_Float16)v;
        }
    }
}

// ---------------------------------------------------------------------------
// pred_curr/pred_warp [..., 1::4, 1::4] -> pcpw (2,2,128,240) and channels
// 126/127 of the 128-channel "inp" concat buffer.  (f16 activations)
// ---------------------------------------------------------------------------
__global__ void downsample_preds(const float* __restrict__ pc_hi,
                                 const float* __restrict__ pw_hi,
                                 _Float16* __restrict__ pcpw,
                                 _Float16* __restrict__ inp128)
{
    int idx = blockIdx.x * blockDim.x + threadIdx.x;
    if (idx >= NB * LOW_HW) return;
    int b = idx / LOW_HW;
    int r = idx - b * LOW_HW;
    int y = r / LOW_W, x = r - (r / LOW_W) * LOW_W;
    long src = (long)b * HI_HW + (long)(1 + 4 * y) * HI_W + (1 + 4 * x);
    float pc = pc_hi[src], pw = pw_hi[src];
    pcpw[(long)b * 2 * LOW_HW + r]          = (_Float16)pc;
    pcpw[(long)b * 2 * LOW_HW + LOW_HW + r] = (_Float16)pw;
    inp128[(long)b * 128 * LOW_HW + 126L * LOW_HW + r] = (_Float16)pc;
    inp128[(long)b * 128 * LOW_HW + 127L * LOW_HW + r] = (_Float16)pw;
}

// ---------------------------------------------------------------------------
// 6 disparity-warped L1 cost planes -> corr_feat channels 25..30 (f16 out).
// Bilinear taps: unconditional clamped loads + validity folded into weights.
// ---------------------------------------------------------------------------
__global__ void cost_volume(const float* __restrict__ feaL,
                            const float* __restrict__ feaR,
                            const _Float16* __restrict__ pcpw,
                            _Float16* __restrict__ corr_feat)
{
    int idx = blockIdx.x * blockDim.x + threadIdx.x;
    if (idx >= NB * LOW_HW) return;
    int b = idx / LOW_HW;
    int r = idx - b * LOW_HW;
    int y = r / LOW_W, x = r - (r / LOW_W) * LOW_W;

    float pc = (float)pcpw[(long)b * 2 * LOW_HW + r];
    float pw = (float)pcpw[(long)b * 2 * LOW_HW + LOW_HW + r];

    int   iA[6], iB[6];
    float w1m[6], w0m[6];
#pragma unroll
    for (int j = 0; j < 6; ++j) {
        float d   = ((j < 3) ? pc : pw) * 0.25f + (float)((j % 3) - 1);
        float src = (float)x - d;
        float f   = floorf(src);
        float a   = src - f;
        int   i0  = (int)f;
        int   i1  = i0 + 1;
        w0m[j] = (i0 >= 0 && i0 < LOW_W) ? (1.f - a) : 0.f;
        w1m[j] = (i1 >= 0 && i1 < LOW_W) ? a : 0.f;
        iA[j]  = i0 < 0 ? 0 : (i0 >= LOW_W ? LOW_W - 1 : i0);
        iB[j]  = i1 < 0 ? 0 : (i1 >= LOW_W ? LOW_W - 1 : i1);
    }
    float accv[6] = {0.f, 0.f, 0.f, 0.f, 0.f, 0.f};
    const float* L = feaL + (long)b * 96 * LOW_HW + (long)y * LOW_W;
    const float* R = feaR + (long)b * 96 * LOW_HW + (long)y * LOW_W;
    for (int c = 0; c < 96; ++c) {
        float        l  = L[(long)c * LOW_HW + x];
        const float* Rc = R + (long)c * LOW_HW;
#pragma unroll
        for (int j = 0; j < 6; ++j) {
            float wv = Rc[iA[j]] * w0m[j] + Rc[iB[j]] * w1m[j];
            accv[j] += fabsf(l - wv);
        }
    }
    long base = (long)b * 31 * LOW_HW + r;
#pragma unroll
    for (int j = 0; j < 6; ++j)
        corr_feat[base + (long)(25 + j) * LOW_HW] = (_Float16)(accv[j] * 0.25f);
}

// ---------------------------------------------------------------------------
// fca (9 planes, ch 0..8) and fsa (16 planes, ch 9..24) of corr_feat (f16).
// featC lives in the 128-channel f16 inp buffer (channels 0..63).
// ---------------------------------------------------------------------------
__global__ void patch_corr(const _Float16* __restrict__ featC,
                           const float* __restrict__ featW,
                           _Float16* __restrict__ corr_feat)
{
    int idx = blockIdx.x * blockDim.x + threadIdx.x;
    if (idx >= NB * LOW_HW) return;
    int b = idx / LOW_HW;
    int r = idx - b * LOW_HW;
    int y = r / LOW_W, x = r - (r / LOW_W) * LOW_W;

    const _Float16* FC   = featC + (long)b * 128 * LOW_HW;
    const float*    FW   = featW + (long)b * 64 * LOW_HW;
    long            outB = (long)b * 31 * LOW_HW + r;

    int q = 0;
    for (int p = 0; p < 9; ++p) {
        int dy = (p / 3 - 1) * 2, dx = (p % 3 - 1) * 2;
        int yy = y + dy, xx = x + dx;
        bool  ok = (yy >= 0 && yy < LOW_H && xx >= 0 && xx < LOW_W);
        int   rs = ok ? (yy * LOW_W + xx) : r;     // clamped, always valid
        float m  = ok ? 1.f : 0.f;
        float sCA = 0.f, sCC = 0.f, sWW = 0.f;
        for (int c = 0; c < 64; ++c) {
            float fc  = (float)FC[(long)c * LOW_HW + r];
            float fw  = FW[(long)c * LOW_HW + r];
            float fcs = (float)FC[(long)c * LOW_HW + rs];
            float fws = FW[(long)c * LOW_HW + rs];
            sCA += fc * fws;
            sCC += fc * fcs;
            sWW += fw * fws;
        }
        corr_feat[outB + (long)p * LOW_HW] = (_Float16)(sCA * m * 0.125f);
        if (p != 4) {
            corr_feat[outB + (long)(9 + q) * LOW_HW]  = (_Float16)(sCC * m * 0.125f);
            corr_feat[outB + (long)(17 + q) * LOW_HW] = (_Float16)(sWW * m * 0.125f);
            ++q;
        }
    }
}

// 4x nearest upsample of fw (f16) into output channel 0 (f32)
__global__ void upsample_fw(const _Float16* __restrict__ fwlow, float* __restrict__ out)
{
    int idx = blockIdx.x * blockDim.x + threadIdx.x;
    if (idx >= NB * HI_HW) return;
    int b = idx / HI_HW;
    int r = idx - b * HI_HW;
    int y = r / HI_W, x = r - (r / HI_W) * HI_W;
    out[(long)b * 2 * HI_HW + r] =
        (float)fwlow[(long)b * LOW_HW + (long)(y >> 2) * LOW_W + (x >> 2)];
}

// ---------------------------------------------------------------------------
// Fully fused high-res forget branch:
//   corr_feat_fr (32 ch, built on the fly) -> 1x1(32->16) into LDS tile with
//   halo -> 3x3(16->8) -> 1x1(8->1) -> sigmoid -> output channel 1.
// ---------------------------------------------------------------------------
__global__ void __launch_bounds__(256)
forget_fused(const float* __restrict__ predC, const float* __restrict__ predW,
             const float* __restrict__ flow,  const float* __restrict__ conf,
             const float* __restrict__ w0, const float* __restrict__ b0,
             const float* __restrict__ w1, const float* __restrict__ b1,
             const float* __restrict__ w2, const float* __restrict__ b2,
             float* __restrict__ out)
{
    __shared__ float sw0[16 * 32];
    __shared__ float sb0[16];
    __shared__ float sw1[8 * 16 * 9];
    __shared__ float sb1[8];
    __shared__ float sw2[8];
    __shared__ float sb2;
    __shared__ float t0[18 * 18 * 16];

    const int tx = threadIdx.x, ty = threadIdx.y;
    const int tid = ty * 16 + tx;
    for (int i = tid; i < 512; i += 256) sw0[i] = w0[i];
    for (int i = tid; i < 1152; i += 256) sw1[i] = w1[i];
    if (tid < 16)      sb0[tid]      = b0[tid];
    else if (tid < 24) sb1[tid - 16] = b1[tid - 16];
    else if (tid < 32) sw2[tid - 24] = w2[tid - 24];
    else if (tid == 32) sb2 = b2[0];
    __syncthreads();

    const int b   = blockIdx.z;
    const int bx0 = blockIdx.x * 16;
    const int by0 = blockIdx.y * 16;
    const float* PC = predC + (long)b * HI_HW;
    const float* PW = predW + (long)b * HI_HW;
    const float* FL = flow + (long)b * 3 * HI_HW;
    const float* CF = conf + (long)b * 3 * HI_HW;

    // Stage t0 = conv1x1(corr_feat_fr) over the 18x18 halo tile
    for (int hidx = tid; hidx < 18 * 18; hidx += 256) {
        int hy = by0 + hidx / 18 - 1;
        int hx = bx0 + hidx % 18 - 1;
        float tloc[16];
        if (hy >= 0 && hy < HI_H && hx >= 0 && hx < HI_W) {
            float ch[32];
            long  cpos = (long)hy * HI_W + hx;
            float pcv = PC[cpos];
            float pwv = PW[cpos];
            int   q = 0;
#pragma unroll
            for (int p = 0; p < 9; ++p) {
                int dy = (p / 3 - 1) * 2, dx = (p % 3 - 1) * 2;
                int yy = hy + dy, xx = hx + dx;
                bool ok  = (yy >= 0 && yy < HI_H && xx >= 0 && xx < HI_W);
                int  yyc = yy < 0 ? 0 : (yy >= HI_H ? HI_H - 1 : yy);
                int  xxc = xx < 0 ? 0 : (xx >= HI_W ? HI_W - 1 : xx);
                long spos = (long)yyc * HI_W + xxc;
                float m   = ok ? 1.f : 0.f;
                float pcs = PC[spos] * m;                 // zero-pad semantics
                float pws = PW[spos] * m;
                ch[p] = fabsf(pcv - pws);                 // dca (9)
                if (p != 4) {
                    ch[9 + q]  = fabsf(pcv - pcs);        // dsa curr (8)
                    ch[17 + q] = fabsf(pwv - pws);        // dsa warp (8)
                    ++q;
                }
            }
            ch[25] = FL[cpos];
            ch[26] = FL[HI_HW + cpos];
            ch[27] = FL[2L * HI_HW + cpos];
            ch[28] = (pwv > 0.f) ? 1.f : 0.f;
            ch[29] = CF[cpos];
            ch[30] = CF[HI_HW + cpos];
            ch[31] = CF[2L * HI_HW + cpos];
#pragma unroll
            for (int oc = 0; oc < 16; ++oc) {
                float t = sb0[oc];
#pragma unroll
                for (int ic = 0; ic < 32; ++ic) t += sw0[oc * 32 + ic] * ch[ic];
                tloc[oc] = t;
            }
        } else {
#pragma unroll
            for (int oc = 0; oc < 16; ++oc) tloc[oc] = 0.f; // zero pad
        }
#pragma unroll
        for (int oc = 0; oc < 16; ++oc) t0[hidx * 16 + oc] = tloc[oc];
    }
    __syncthreads();

    int y = by0 + ty, x = bx0 + tx;
    if (y >= HI_H || x >= HI_W) return;
    float t1[8];
#pragma unroll
    for (int oc = 0; oc < 8; ++oc) t1[oc] = sb1[oc];
#pragma unroll
    for (int ky = 0; ky < 3; ++ky)
#pragma unroll
        for (int kx = 0; kx < 3; ++kx) {
            int pos = (ty + ky) * 18 + (tx + kx);
#pragma unroll
            for (int ic = 0; ic < 16; ++ic) {
                float v = t0[pos * 16 + ic];
#pragma unroll
                for (int oc = 0; oc < 8; ++oc)
                    t1[oc] += sw1[(oc * 16 + ic) * 9 + ky * 3 + kx] * v;
            }
        }
    float s = sb2;
#pragma unroll
    for (int oc = 0; oc < 8; ++oc) s += sw2[oc] * t1[oc];
    out[(long)b * 2 * HI_HW + HI_HW + (long)y * HI_W + x] = 1.f / (1.f + __expf(-s));
}

// ---------------------------------------------------------------------------
extern "C" void kernel_launch(void* const* d_in, const int* in_sizes, int n_in,
                              void* d_out, int out_size, void* d_ws, size_t ws_size,
                              hipStream_t stream)
{
    (void)in_sizes; (void)n_in; (void)out_size; (void)ws_size;

    const float* left  = (const float*)d_in[0];
    const float* feaR  = (const float*)d_in[1];
    const float* featW = (const float*)d_in[2];
    const float* predC = (const float*)d_in[3];
    const float* predW = (const float*)d_in[4];
    const float* flow  = (const float*)d_in[5];
    const float* conf  = (const float*)d_in[6];
    const float* W_[12], *B_[12];
    W_[0]  = (const float*)d_in[7];  B_[0]  = (const float*)d_in[8];   // kl0
    W_[1]  = (const float*)d_in[9];  B_[1]  = (const float*)d_in[10];  // kl1
    W_[2]  = (const float*)d_in[11]; B_[2]  = (const float*)d_in[12];  // kl2
    W_[3]  = (const float*)d_in[13]; B_[3]  = (const float*)d_in[14];  // kl3
    W_[4]  = (const float*)d_in[15]; B_[4]  = (const float*)d_in[16];  // cc0
    W_[5]  = (const float*)d_in[17]; B_[5]  = (const float*)d_in[18];  // cc1
    W_[6]  = (const float*)d_in[19]; B_[6]  = (const float*)d_in[20];  // cd0
    W_[7]  = (const float*)d_in[21]; B_[7]  = (const float*)d_in[22];  // cd1
    W_[8]  = (const float*)d_in[23]; B_[8]  = (const float*)d_in[24];  // mc
    W_[9]  = (const float*)d_in[25]; B_[9]  = (const float*)d_in[26];  // wh0
    W_[10] = (const float*)d_in[27]; B_[10] = (const float*)d_in[28];  // wh1
    W_[11] = (const float*)d_in[35]; B_[11] = (const float*)d_in[36];  // rc
    const float* fh_w0 = (const float*)d_in[29]; const float* fh_b0 = (const float*)d_in[30];
    const float* fh_w1 = (const float*)d_in[31]; const float* fh_b1 = (const float*)d_in[32];
    const float* fh_w2 = (const float*)d_in[33]; const float* fh_b2 = (const float*)d_in[34];

    // ---- f16 activation workspace ----
    _Float16*  ws  = (_Float16*)d_ws;
    const long C64 = (long)NB * 64 * LOW_HW;
    _Float16* leftH  = ws;                                    // left_feat as f16
    _Float16* bufA0  = leftH + (long)NB * 96 * LOW_HW;        // x -> disp1 -> wh hidden
    _Float16* bufR1  = bufA0 + C64;                           // r1 -> cc hidden lo -> net
    _Float16* bufR2  = bufR1 + C64;                           // r2 -> cc hidden hi
    _Float16* inp128 = bufR2 + C64;                           // feat_curr|mo|pc|pw
    _Float16* corrF  = inp128 + (long)NB * 128 * LOW_HW;      // 31-ch corr_feat
    _Float16* cd128  = corrF + (long)NB * 31 * LOW_HW;        // corr|disp
    _Float16* pcpw   = cd128 + (long)NB * 128 * LOW_HW;       // (2,2,H,W)
    _Float16* fwlow  = pcpw + (long)NB * 2 * LOW_HW;          // (2,1,H,W)
    _Float16* wsEnd  = fwlow + (long)NB * LOW_HW;

    // ---- packed-weight area (f16): {Cout, Cin, KS}, Cout padded to 32 ----
    const int CO[12]  = {64, 64, 64, 64, 128, 64, 64, 64, 62, 64, 1, 64};
    const int CI[12]  = {96, 64, 64, 64, 31, 128, 2, 64, 128, 64, 64, 128};
    const int KSZ[12] = {1, 3, 3, 1, 1, 1, 7, 3, 7, 3, 1, 3};
    _Float16* wp[12];
    int nmt[12], nkt[12];
    {
        _Float16* cur = wsEnd;
        for (int i = 0; i < 12; ++i) {
            int Ktot = CI[i] * KSZ[i] * KSZ[i];
            nmt[i] = 2 * ((CO[i] + 31) / 32);   // even # of 16-row tiles
            nkt[i] = (Ktot + 31) / 32;
            wp[i]  = cur;
            cur   += (long)nmt[i] * nkt[i] * 512;
        }
        for (int i = 0; i < 12; ++i) {
            int Ktot  = CI[i] * KSZ[i] * KSZ[i];
            int total = nmt[i] * nkt[i] * 512;
            pack_weights<<<(total + 255) / 256, 256, 0, stream>>>(
                W_[i], wp[i], CO[i], Ktot, nmt[i], nkt[i]);
        }
    }

    // cast left_feat once
    {
        int n = NB * 96 * LOW_HW;
        cast_f32_f16<<<(n + 255) / 256, 256, 0, stream>>>(left, leftH, n);
    }

    const dim3 blk(128);
    const int  NTX = (LOW_HW + 63) / 64;   // 480 (4 waves x 16 px per block)
    const long BS64  = (long)64 * LOW_HW;
    const long BS96  = (long)96 * LOW_HW;
    const long BS128 = (long)128 * LOW_HW;
    const long BS31  = (long)31 * LOW_HW;
    const long BS2   = (long)2 * LOW_HW;
    const long BS1   = (long)LOW_HW;

    // ---- feature head (kl) ----
    conv2d_wmma<1><<<dim3(NTX, nmt[0] / 2, NB), blk, 0, stream>>>(leftH, 96, BS96,
        wp[0], nkt[0], B_[0], bufA0, 64, BS64, 0, nullptr, 0,
        LOW_H, LOW_W, 0, ACT_RELU, 0);
    conv2d_wmma<3><<<dim3(NTX, nmt[1] / 2, NB), blk, 0, stream>>>(bufA0, 64, BS64,
        wp[1], nkt[1], B_[1], bufR1, 64, BS64, 0, nullptr, 0,
        LOW_H, LOW_W, 1, ACT_MISH, 0);
    conv2d_wmma<3><<<dim3(NTX, nmt[2] / 2, NB), blk, 0, stream>>>(bufR1, 64, BS64,
        wp[2], nkt[2], B_[2], bufR2, 64, BS64, 0, bufA0, BS64,
        LOW_H, LOW_W, 1, ACT_NONE, 0);
    conv2d_wmma<1><<<dim3(NTX, nmt[3] / 2, NB), blk, 0, stream>>>(bufR2, 64, BS64,
        wp[3], nkt[3], B_[3], inp128, 64, BS128, 0, nullptr, 0,
        LOW_H, LOW_W, 0, ACT_NONE, 1);   // feat_curr = conv(relu(r))

    // ---- downsampled preds, cost volumes, patch correlations ----
    downsample_preds<<<(NB * LOW_HW + 255) / 256, 256, 0, stream>>>(predC, predW, pcpw, inp128);
    cost_volume<<<(NB * LOW_HW + 255) / 256, 256, 0, stream>>>(left, feaR, pcpw, corrF);
    patch_corr<<<(NB * LOW_HW + 255) / 256, 256, 0, stream>>>(inp128, featW, corrF);

    // ---- cc: corr = relu(1x1(relu(1x1(corr_feat)))) -> cd128 ch0..63 ----
    conv2d_wmma<1><<<dim3(NTX, nmt[4] / 2, NB), blk, 0, stream>>>(corrF, 31, BS31,
        wp[4], nkt[4], B_[4], bufR1, 128, BS128, 0, nullptr, 0,
        LOW_H, LOW_W, 0, ACT_RELU, 0);
    conv2d_wmma<1><<<dim3(NTX, nmt[5] / 2, NB), blk, 0, stream>>>(bufR1, 128, BS128,
        wp[5], nkt[5], B_[5], cd128, 64, BS128, 0, nullptr, 0,
        LOW_H, LOW_W, 0, ACT_RELU, 0);

    // ---- cd: disp branch -> cd128 ch64..127 ----
    conv2d_wmma<7><<<dim3(NTX, nmt[6] / 2, NB), blk, 0, stream>>>(pcpw, 2, BS2,
        wp[6], nkt[6], B_[6], bufA0, 64, BS64, 0, nullptr, 0,
        LOW_H, LOW_W, 3, ACT_RELU, 0);
    conv2d_wmma<3><<<dim3(NTX, nmt[7] / 2, NB), blk, 0, stream>>>(bufA0, 64, BS64,
        wp[7], nkt[7], B_[7], cd128, 64, BS128, 64, nullptr, 0,
        LOW_H, LOW_W, 1, ACT_RELU, 0);

    // ---- mc: mo = relu(7x7(cat(corr,disp))) -> inp128 ch64..125 ----
    conv2d_wmma<7><<<dim3(NTX, nmt[8] / 2, NB), blk, 0, stream>>>(cd128, 128, BS128,
        wp[8], nkt[8], B_[8], inp128, 62, BS128, 64, nullptr, 0,
        LOW_H, LOW_W, 3, ACT_RELU, 0);

    // ---- rc: net = relu(3x3(inp)) + corr ----
    conv2d_wmma<3><<<dim3(NTX, nmt[11] / 2, NB), blk, 0, stream>>>(inp128, 128, BS128,
        wp[11], nkt[11], B_[11], bufR1, 64, BS64, 0, cd128, BS128,
        LOW_H, LOW_W, 1, ACT_RELU, 0);

    // ---- wh: fw = sigmoid(1x1(3x3(net))) ----
    conv2d_wmma<3><<<dim3(NTX, nmt[9] / 2, NB), blk, 0, stream>>>(bufR1, 64, BS64,
        wp[9], nkt[9], B_[9], bufA0, 64, BS64, 0, nullptr, 0,
        LOW_H, LOW_W, 1, ACT_NONE, 0);
    conv2d_wmma<1><<<dim3(NTX, nmt[10] / 2, NB), blk, 0, stream>>>(bufA0, 64, BS64,
        wp[10], nkt[10], B_[10], fwlow, 1, BS1, 0, nullptr, 0,
        LOW_H, LOW_W, 0, ACT_SIGMOID, 0);

    // ---- outputs ----
    upsample_fw<<<(NB * HI_HW + 255) / 256, 256, 0, stream>>>(fwlow, (float*)d_out);
    forget_fused<<<dim3(HI_W / 16, HI_H / 16, NB), dim3(16, 16), 0, stream>>>(
        predC, predW, flow, conf, fh_w0, fh_b0, fh_w1, fh_b1, fh_w2, fh_b2,
        (float*)d_out);
}
